// GCN_41360535061036
// MI455X (gfx1250) — compile-verified, matrix operand order checked by reading, not run on hardware
//
#include <hip/hip_runtime.h>

typedef __attribute__((ext_vector_type(16))) __bf16 v16bf;
typedef __attribute__((ext_vector_type(8)))  float  v8f;

// ---------------- normalization kernels ----------------

__global__ void k_fill1(float* p, int n) {
  int i = blockIdx.x * blockDim.x + threadIdx.x;
  if (i < n) p[i] = 1.0f;                       // self-loop weight contributes 1 to degree
}

__global__ void k_deg(const int* __restrict__ col, const float* __restrict__ w,
                      float* deg, int nnz) {
  int e = blockIdx.x * blockDim.x + threadIdx.x;
  if (e < nnz) atomicAdd(&deg[col[e]], w[e]);
}

__global__ void k_rsqrt(float* d, int n) {
  int i = blockIdx.x * blockDim.x + threadIdx.x;
  if (i < n) { float v = d[i]; d[i] = v > 0.f ? rsqrtf(v) : 0.f; }
}

__global__ void k_norm(const int* __restrict__ row, const int* __restrict__ col,
                       const float* __restrict__ w, const float* __restrict__ dis,
                       float* __restrict__ nrm, int nnz) {
  int e = blockIdx.x * blockDim.x + threadIdx.x;
  if (e < nnz) nrm[e] = dis[row[e]] * w[e] * dis[col[e]];
}

// ---------------- B-matrix fragment packing ----------------
// Convert W (K x NB, f32, row-major) into bf16 WMMA B-fragments laid out as
// [ntile][kstep][lane][16] so the GEMM loads one contiguous 32B vector per
// lane per k-step. Zero-pads columns n >= NB (C=40 tail). Runs once per call
// on a handful of blocks; divergence here is irrelevant.

template<int K, int NB>
__global__ void __launch_bounds__(32)
k_pack_b(const float* __restrict__ B, __bf16* __restrict__ Bp) {
  const int lane = threadIdx.x & 31;
  const int half = lane >> 4;
  const int l    = lane & 15;
  const int ksteps = K / 32;
  const int kk = blockIdx.x % ksteps;
  const int nt = blockIdx.x / ksteps;
  const int n  = nt * 16 + l;
  const int bk = kk * 32 + half * 8;
  __bf16* dst = Bp + ((size_t)blockIdx.x * 32 + lane) * 16;
  #pragma unroll
  for (int i = 0; i < 8; ++i) {
    float blo = (n < NB) ? B[(size_t)(bk + i)      * NB + n] : 0.f;
    float bhi = (n < NB) ? B[(size_t)(bk + 16 + i) * NB + n] : 0.f;
    dst[i]     = (__bf16)blo;
    dst[8 + i] = (__bf16)bhi;
  }
}

// ---------------- WMMA GEMM: C[M x NB] = A[M x K] * B[K x NB] ----------------
// One wave32 per 16x16 output tile. f32 A converted to bf16 in-flight (four
// B128 loads + v_cvt_pk_bf16_f32 per k-step); B fragment is a single aligned
// 32B vector load from the pre-packed buffer. f32 accumulate via
// v_wmma_f32_16x16x32_bf16.

template<int K, int NB>
__global__ void __launch_bounds__(32)
k_gemm_wmma(const float* __restrict__ A, const __bf16* __restrict__ Bp,
            float* __restrict__ C, int M) {
  const int lane = threadIdx.x & 31;
  const int half = lane >> 4;          // 0: lanes 0-15, 1: lanes 16-31
  const int l    = lane & 15;
  const int m0   = blockIdx.x << 4;
  const int n0   = blockIdx.y << 4;
  int m = m0 + l; if (m >= M) m = 0;   // clamp (branch-free); garbage rows never stored
  const int n = n0 + l;

  constexpr int ksteps = K / 32;
  const v16bf* bfrag =
      (const v16bf*)(Bp + (size_t)blockIdx.y * ksteps * 32 * 16);

  v8f acc = {};
  #pragma unroll
  for (int kk = 0; kk < ksteps; ++kk) {
    const int bk = kk * 32 + half * 8;
    const float4* ap = reinterpret_cast<const float4*>(A + (size_t)m * K + bk);
    float4 a0 = ap[0], a1 = ap[1], a2 = ap[4], a3 = ap[5];
    v16bf a;
    a[0]=(__bf16)a0.x;  a[1]=(__bf16)a0.y;  a[2]=(__bf16)a0.z;  a[3]=(__bf16)a0.w;
    a[4]=(__bf16)a1.x;  a[5]=(__bf16)a1.y;  a[6]=(__bf16)a1.z;  a[7]=(__bf16)a1.w;
    a[8]=(__bf16)a2.x;  a[9]=(__bf16)a2.y;  a[10]=(__bf16)a2.z; a[11]=(__bf16)a2.w;
    a[12]=(__bf16)a3.x; a[13]=(__bf16)a3.y; a[14]=(__bf16)a3.z; a[15]=(__bf16)a3.w;
    v16bf b = bfrag[kk * 32 + lane];
    acc = __builtin_amdgcn_wmma_f32_16x16x32_bf16(
        /*neg_a=*/false, a, /*neg_b=*/false, b,
        /*c_mod=*/(short)0, acc, /*reuse_a=*/false, /*reuse_b=*/false);
  }
  #pragma unroll
  for (int r = 0; r < 8; ++r) {
    const int mm = m0 + (half << 3) + r;
    if (mm < M && n < NB) C[(size_t)mm * NB + n] = acc[r];
  }
}

// ---------------- aggregation kernels ----------------

// agg[i,f] = dis[i]^2 * t[i,f]   (self-loop contribution, norm_self = dis^2)
__global__ void k_self_init(const float* __restrict__ t, const float* __restrict__ dis,
                            float* __restrict__ agg, int N, int F) {
  long long g = (long long)blockIdx.x * blockDim.x + threadIdx.x;
  long long tot = (long long)N * F;
  if (g >= tot) return;
  int i = (int)(g / F);
  float s = dis[i];
  agg[g] = s * s * t[g];
}

// agg[col[e], f] += norm[e] * t[row[e], f]
template<int F>
__global__ void k_scatter(const int* __restrict__ row, const int* __restrict__ col,
                          const float* __restrict__ nrm, const float* __restrict__ t,
                          float* __restrict__ agg, int nnz) {
  long long g = (long long)blockIdx.x * blockDim.x + threadIdx.x;
  long long tot = (long long)nnz * F;
  if (g >= tot) return;
  int e = (int)(g / F);
  int f = (int)(g % F);
  float v = nrm[e] * t[(size_t)row[e] * F + f];
  atomicAdd(&agg[(size_t)col[e] * F + f], v);
}

__global__ void k_relu_bias(float* __restrict__ h, const float* __restrict__ bias,
                            long long tot, int F) {
  long long g = (long long)blockIdx.x * blockDim.x + threadIdx.x;
  if (g >= tot) return;
  int f = (int)(g % F);
  float v = h[g] + bias[f];
  h[g] = v > 0.f ? v : 0.f;
}

// out[i,f] = b2[f] + dis[i]^2 * t2[i,f]
__global__ void k_out_init(const float* __restrict__ t2, const float* __restrict__ dis,
                           const float* __restrict__ bias, float* __restrict__ out,
                           int N, int F) {
  long long g = (long long)blockIdx.x * blockDim.x + threadIdx.x;
  long long tot = (long long)N * F;
  if (g >= tot) return;
  int i = (int)(g / F);
  int f = (int)(g % F);
  float s = dis[i];
  out[g] = bias[f] + s * s * t2[g];
}

// ---------------- launcher ----------------

extern "C" void kernel_launch(void* const* d_in, const int* in_sizes, int n_in,
                              void* d_out, int out_size, void* d_ws, size_t ws_size,
                              hipStream_t stream) {
  (void)n_in; (void)out_size; (void)ws_size;

  const float* x  = (const float*)d_in[0];   // N x 128
  const int*   ei = (const int*)  d_in[1];   // 2 x NNZ
  const float* ew = (const float*)d_in[2];   // NNZ
  const float* W1 = (const float*)d_in[3];   // 128 x 64
  const float* b1 = (const float*)d_in[4];   // 64
  const float* W2 = (const float*)d_in[5];   // 64 x 40
  const float* b2 = (const float*)d_in[6];   // 40
  float* out = (float*)d_out;                // N x 40

  const int D = 128;
  const int H = 64;
  const int Ccls = 40;
  const int N   = in_sizes[0] / D;
  const int NNZ = in_sizes[2];
  const int* row = ei;          // sources
  const int* col = ei + NNZ;    // targets

  // workspace layout (floats): dis[N] | norm[NNZ] | t[N*H] | agg[N*H] | bf16 packs
  float* dis  = (float*)d_ws;
  float* nrm  = dis + N;
  float* t    = nrm + NNZ;                 // GEMM1 out (N*64), later GEMM2 out (N*40)
  float* agg  = t + (size_t)N * H;         // layer-1 aggregate -> ReLU'd hidden
  __bf16* bp1 = (__bf16*)(agg + (size_t)N * H);  // 4 ntiles * 4 ksteps * 32 * 16
  __bf16* bp2 = bp1 + 4 * 4 * 32 * 16;           // 3 ntiles * 2 ksteps * 32 * 16

  auto nblk = [](long long tot, int tpb) { return (unsigned)((tot + tpb - 1) / tpb); };

  // --- pack weight matrices into WMMA bf16 fragment layout (tiny, once) ---
  k_pack_b<128, 64><<<4 * 4, 32, 0, stream>>>(W1, bp1);
  k_pack_b<64, 40><<<3 * 2, 32, 0, stream>>>(W2, bp2);

  // --- normalization (deg uses dis buffer in place) ---
  k_fill1<<<nblk(N, 256), 256, 0, stream>>>(dis, N);
  k_deg  <<<nblk(NNZ, 256), 256, 0, stream>>>(col, ew, dis, NNZ);
  k_rsqrt<<<nblk(N, 256), 256, 0, stream>>>(dis, N);
  k_norm <<<nblk(NNZ, 256), 256, 0, stream>>>(row, col, ew, dis, nrm, NNZ);

  // --- layer 1: t = x @ W1 (bf16 WMMA) ---
  dim3 g1((N + 15) / 16, H / 16);
  k_gemm_wmma<128, 64><<<g1, 32, 0, stream>>>(x, bp1, t, N);

  long long totH = (long long)N * H;
  k_self_init<<<nblk(totH, 256), 256, 0, stream>>>(t, dis, agg, N, H);
  k_scatter<64><<<nblk((long long)NNZ * 64, 256), 256, 0, stream>>>(row, col, nrm, t, agg, NNZ);
  k_relu_bias<<<nblk(totH, 256), 256, 0, stream>>>(agg, b1, totH, H);

  // --- layer 2: t = h @ W2 (bf16 WMMA, N padded to 3 tiles) ---
  dim3 g2((N + 15) / 16, (Ccls + 15) / 16);
  k_gemm_wmma<64, 40><<<g2, 32, 0, stream>>>(agg, bp2, t, N);

  long long totC = (long long)N * Ccls;
  k_out_init<<<nblk(totC, 256), 256, 0, stream>>>(t, dis, b2, out, N, Ccls);
  k_scatter<40><<<nblk((long long)NNZ * 40, 256), 256, 0, stream>>>(row, col, nrm, t, out, NNZ);
}
